// SamplingLayer_4492535792606
// MI455X (gfx1250) — matrix operand order repaired and checked by reading, not compiled, not run on hardware
//
#include <hip/hip_runtime.h>

typedef __attribute__((ext_vector_type(16))) _Float16 v16h;
typedef __attribute__((ext_vector_type(8)))  float    v8f;

#define BATCH  8
#define NC     8
#define HH     512
#define WW     512
#define NSTEPS 10

__device__ __forceinline__ unsigned int fmix32(unsigned int x) {
    x ^= x >> 16; x *= 0x7feb352du;
    x ^= x >> 15; x *= 0x846ca68bu;
    x ^= x >> 16;
    return x;
}

// One synchronous Gibbs sweep. Block = 256 threads (8 waves); each wave owns
// 32 consecutive pixels of one image row. Per wave: 2x v_wmma_f32_16x16x32_f16
// compute the negated 4-neighbor pairwise energies for 16 pixels each
// (B = one-hot neighbor labels, K = 4 neighbors * 8 labels = 32).
__global__ __launch_bounds__(256) void gibbs_step_kernel(
    const float* __restrict__ unaries,   // [B,C,H,W]
    const float* __restrict__ binaries,  // [4,C,C]
    const int*   __restrict__ s_in,      // [B,H,W]
    int*         __restrict__ s_out,     // [B,H,W]
    float*       __restrict__ prob_acc,  // [B,C,H,W] running marginal
    int*         __restrict__ s_final,   // [B,H,W] written on last step
    int step, int last)
{
    // [wave][class row 0..15][pixel-in-wave] : 16 KB. Rows 8..15 are the
    // zero rows produced by lanes 16..31 -- written (branch-free) but unread.
    __shared__ float smem[8][16][32];

    const int bid  = blockIdx.x;
    const int b    = bid >> 10;              // B*H*2 blocks: b = bid/1024
    const int h    = (bid >> 1) & (HH - 1);
    const int x0   = (bid & 1) * 256;
    const int tid  = threadIdx.x;
    const int wave = tid >> 5;
    const int lane = tid & 31;
    const int grp  = lane >> 4;              // lane group (0/1) for WMMA layouts
    const int col  = lane & 15;              // matrix column within tile
    const int base = x0 + wave * 32;
    const int x    = base + lane;            // pixel owned in scalar phase
    const int rowBase = (b * HH + h) * WW;

    // Warm L2/L0 for the scalar-phase unary row (emits global_prefetch_b8).
    __builtin_prefetch(&unaries[((b * NC) * HH + h) * WW + x], 0, 3);

    // ---- A matrix: 16x32 f16 holding -binaries (sign folded in, so
    // neg_a stays false). Rows m = class (m>=8 zero), k = neighbor*8+label.
    // 16-bit A layout: lane m=lane&15; VGPR v half hh:
    //   k = (v<4 ? grp*8 + 2v+hh : 16 + grp*8 + 2(v-4)+hh)
    v16h amat;
#pragma unroll
    for (int v = 0; v < 8; ++v) {
#pragma unroll
        for (int hh = 0; hh < 2; ++hh) {
            int k = (v < 4) ? (grp * 8 + v * 2 + hh)
                            : (16 + grp * 8 + (v - 4) * 2 + hh);
            float val = (col < NC)
                      ? -binaries[(k >> 3) * (NC * NC) + col * NC + (k & 7)]
                      : 0.0f;
            amat[v * 2 + hh] = (_Float16)val;
        }
    }

    // ---- Two 16-pixel WMMA tiles per wave.
#pragma unroll
    for (int t = 0; t < 2; ++t) {
        const int px = base + t * 16 + col;
        const int gi = rowBase + px;

        // Branch-free neighbor fetch: clamp the address so the load is always
        // in-bounds and unconditional, then select label = -1 (mask folded in)
        // for invalid neighbors afterwards (plain v_cndmask, no exec branches).
        const bool okR = (px + 1 < WW);
        const bool okD = (h  + 1 < HH);
        const bool okL = (px > 0);
        const bool okU = (h  > 0);
        const int vR = s_in[gi + (okR ? 1  : 0)];
        const int vD = s_in[gi + (okD ? WW : 0)];
        const int vL = s_in[gi - (okL ? 1  : 0)];
        const int vU = s_in[gi - (okU ? WW : 0)];
        int nb[4];
        nb[0] = okR ? vR : -1;   // (0,+1)
        nb[1] = okD ? vD : -1;   // (+1,0)
        nb[2] = okL ? vL : -1;   // (0,-1)
        nb[3] = okU ? vU : -1;   // (-1,0)

        // B matrix 32x16 f16 one-hot: column n = lane&15,
        // VGPR v half hh holds k = (v<4 ? grp*8 + 2v+hh : 16 + grp*8 + 2(v-4)+hh).
        v16h bmat;
#pragma unroll
        for (int v = 0; v < 8; ++v) {
#pragma unroll
            for (int hh = 0; hh < 2; ++hh) {
                int k = (v < 4) ? (grp * 8 + v * 2 + hh)
                                : (16 + grp * 8 + (v - 4) * 2 + hh);
                bmat[v * 2 + hh] = (nb[k >> 3] == (k & 7)) ? (_Float16)1.0f
                                                           : (_Float16)0.0f;
            }
        }

        v8f cacc = {};
        // D = (-binaries)*onehot + 0 -> negated pairwise energy (partial logits).
        v8f d = __builtin_amdgcn_wmma_f32_16x16x32_f16(
            /*neg_a=*/false, amat, /*neg_b=*/false, bmat,
            /*c_mod=*/(short)0, cacc, /*reuse_a=*/false, /*reuse_b=*/false);

        // Branch-free stage-out: lane group g holds rows M = r + 8*g,
        // column = t*16+col. All 32 lanes write disjoint slots.
#pragma unroll
        for (int r = 0; r < NC; ++r)
            smem[wave][r + grp * 8][t * 16 + col] = d[r];
    }
    __syncthreads();

    // ---- Scalar phase: every lane owns pixel x = base+lane.
    float lg[NC];
#pragma unroll
    for (int c = 0; c < NC; ++c)
        lg[c] = smem[wave][c][lane]
              - unaries[((b * NC + c) * HH + h) * WW + x];

    // Softmax over C=8 (logits = -energy).
    float mx = lg[0];
#pragma unroll
    for (int c = 1; c < NC; ++c) mx = fmaxf(mx, lg[c]);
    float ex[NC];
    float sum = 0.0f;
#pragma unroll
    for (int c = 0; c < NC; ++c) { ex[c] = __expf(lg[c] - mx); sum += ex[c]; }
    const float inv = (1.0f / (float)NSTEPS) / sum;

    // Uniform branch: step 0 overwrites (skips the read), later steps RMW.
    if (step == 0) {
#pragma unroll
        for (int c = 0; c < NC; ++c)
            prob_acc[((b * NC + c) * HH + h) * WW + x] = ex[c] * inv;
    } else {
#pragma unroll
        for (int c = 0; c < NC; ++c) {
            const int oi = ((b * NC + c) * HH + h) * WW + x;
            prob_acc[oi] += ex[c] * inv;
        }
    }

    // Gumbel-max categorical draw (counter-based hash RNG).
    const unsigned int gi32 = (unsigned int)(rowBase + x);
    int best = 0;
    float bestv = -3.0e38f;
#pragma unroll
    for (int c = 0; c < NC; ++c) {
        unsigned int hsh = fmix32(gi32 * (unsigned int)NC + (unsigned int)c
                                  + (unsigned int)step * 0x9E3779B9u);
        float u = (float)(hsh >> 9) * (1.0f / 8388608.0f) + 1.0e-7f;
        float gmb = -__logf(-__logf(u));
        float v = lg[c] + gmb;
        if (v > bestv) { bestv = v; best = c; }
    }

    s_out[rowBase + x] = best;
    if (last) s_final[rowBase + x] = best;
}

extern "C" void kernel_launch(void* const* d_in, const int* in_sizes, int n_in,
                              void* d_out, int out_size, void* d_ws, size_t ws_size,
                              hipStream_t stream) {
    (void)in_sizes; (void)n_in; (void)out_size; (void)ws_size;
    const float* unaries  = (const float*)d_in[0];
    const float* binaries = (const float*)d_in[1];
    const int*   sample0  = (const int*)d_in[2];

    float* prob_acc = (float*)d_out;
    int*   s_final  = (int*)((float*)d_out + (size_t)BATCH * NC * HH * WW);

    int* sA = (int*)d_ws;                       // ping-pong sample buffers
    int* sB = sA + (size_t)BATCH * HH * WW;     // needs 2 * 8 MB of d_ws
    int* bufs[2] = { sA, sB };

    const int nblocks = BATCH * HH * 2;         // 8192 blocks x 256 threads
    const int* cur = sample0;
    for (int k = 0; k < NSTEPS; ++k) {
        int* nxt = bufs[k & 1];
        gibbs_step_kernel<<<nblocks, 256, 0, stream>>>(
            unaries, binaries, cur, nxt, prob_acc, s_final,
            k, (k == NSTEPS - 1) ? 1 : 0);
        cur = nxt;
    }
}
